// Model3DSSG_36979668418612
// MI455X (gfx1250) — compile-verified, matrix operand order checked by reading, not run on hardware
//
#include <hip/hip_runtime.h>
#include <hip/hip_bf16.h>

#define NNODES 50000
#define NEDGES 1000000
#define ED     32
#define HID    128
#define NCLS   27          // C + 1

#define MTILES 4           // 16-edge M-tiles per wave (64 edges/wave)
#define EPB    (MTILES*16) // edges per block

typedef __attribute__((ext_vector_type(16))) _Float16 v16h;
typedef __attribute__((ext_vector_type(8)))  float    v8f;
typedef __attribute__((ext_vector_type(4)))  float    f32x4;

union HalfVec { v16h h; f32x4 q[2]; };

// LDS row strides (in halfs): row*stride*2 stays 16B aligned and rows land on
// different banks (784=16*49, 528=16*33, 272=16*17 bytes).
#define EIN_STRIDE 392
#define H1_STRIDE  264
#define H2_STRIDE  136

// dynamic LDS budget: 64*392 + 64*264 halfs = 41984 halfs = 83968 bytes
#define LDS_HALFS  (EPB*EIN_STRIDE + EPB*H1_STRIDE)

// ---------------------------------------------------------------------------
// Kernel 1: per-node embeddings -> subj/obj [N,128] stored as f16
// ---------------------------------------------------------------------------
__global__ void node_embed_kernel(const int* __restrict__ orient,
                                  const float* __restrict__ dim,
                                  const float* __restrict__ otab,
                                  const float* __restrict__ sW,
                                  const float* __restrict__ sb,
                                  const float* __restrict__ subjW,
                                  const float* __restrict__ subjb,
                                  const float* __restrict__ objW,
                                  const float* __restrict__ objb,
                                  _Float16* __restrict__ subj_out,
                                  _Float16* __restrict__ obj_out)
{
    const int n = blockIdx.x * blockDim.x + threadIdx.x;
    if (n >= NNODES) return;

    const float d0 = dim[n*3+0], d1 = dim[n*3+1], d2 = dim[n*3+2];
    const int   o  = orient[n];

    float cat[2*ED];
    #pragma unroll
    for (int j = 0; j < ED; ++j)
        cat[j] = sb[j] + d0*sW[j] + d1*sW[ED+j] + d2*sW[2*ED+j];
    #pragma unroll
    for (int j = 0; j < ED; ++j)
        cat[ED+j] = otab[o*ED + j];

    for (int h = 0; h < HID; ++h) {
        float a = subjb[h];
        float b = objb[h];
        #pragma unroll 8
        for (int j = 0; j < 2*ED; ++j) {
            const float c = cat[j];
            a = fmaf(c, subjW[j*HID + h], a);
            b = fmaf(c, objW[j*HID + h],  b);
        }
        subj_out[(size_t)n*HID + h] = (_Float16)a;
        obj_out [(size_t)n*HID + h] = (_Float16)b;
    }
}

// ---------------------------------------------------------------------------
// Kernel 2: repack a [K, realN] f32 weight into WMMA f16 B-fragments.
// Fragment = 32x16 (KxN) tile, 512 halfs, laid out so lane l reads its 16
// halfs contiguously: n = l&15, kh = l>>4, halfs t<8 -> k = kh*8+t,
// t>=8 -> k = 16 + kh*8 + (t-8).  out index == global thread index.
// ---------------------------------------------------------------------------
__global__ void repack_kernel(const float* __restrict__ W,
                              _Float16* __restrict__ out,
                              int NT, int realN, int total)
{
    const int idx = blockIdx.x * blockDim.x + threadIdx.x;
    if (idx >= total) return;
    const int t    = idx & 15;
    const int l    = (idx >> 4) & 31;
    const int frag = idx >> 9;
    const int kt   = frag / NT;
    const int nt   = frag % NT;
    const int n    = nt*16 + (l & 15);
    const int kh   = l >> 4;
    const int kl   = (t < 8) ? (kh*8 + t) : (16 + kh*8 + (t - 8));
    const int k    = kt*32 + kl;
    const float v  = (n < realN) ? W[k*realN + n] : 0.0f;
    out[idx] = (_Float16)v;
}

// ---------------------------------------------------------------------------
// Kernel 3: per-edge 3-layer MLP with v_wmma_f32_16x16x32_f16.
// One wave per block; each wave owns FOUR 16-edge M-tiles (64 edges) so every
// B fragment is loaded once from L2 and fed to four WMMAs (register-level B
// reuse -> weight traffic ~4 GB total, 4 independent WMMA chains per wave).
// A-panel + h1 live in dynamic LDS (84 KB; CDNA5 allows 320 KB/workgroup).
// ---------------------------------------------------------------------------
__global__ __launch_bounds__(32) void edge_mlp_kernel(
    const int* __restrict__ src, const int* __restrict__ dst,
    const float* __restrict__ loc,
    const float* __restrict__ transW, const float* __restrict__ transb,
    const _Float16* __restrict__ subj, const _Float16* __restrict__ obj,
    const _Float16* __restrict__ w1f, const float* __restrict__ b1,
    const _Float16* __restrict__ w2f, const float* __restrict__ b2,
    const _Float16* __restrict__ w3f, const float* __restrict__ b3,
    float* __restrict__ scores)
{
    extern __shared__ _Float16 smem[];           // dynamic LDS, 83968 bytes
    _Float16* ein = smem;                        // [EPB x EIN_STRIDE], reused as h2
    _Float16* h1  = smem + EPB*EIN_STRIDE;       // [EPB x H1_STRIDE]

    const int lane = threadIdx.x & 31;
    const long edge_base = (long)blockIdx.x * EPB;
    if (edge_base >= NEDGES) return;

    // ---------------- phase 1: build edge_in [EPB x 384] f16 ---------------
    // each lane owns 4 feature columns of each 128-wide sub-block
    float tw0[4], tw1[4], tw2[4], tb[4];
    #pragma unroll
    for (int j = 0; j < 4; ++j) {
        tw0[j] = transW[0*HID + lane*4 + j];
        tw1[j] = transW[1*HID + lane*4 + j];
        tw2[j] = transW[2*HID + lane*4 + j];
        tb[j]  = transb[lane*4 + j];
    }

    for (int i = 0; i < EPB; ++i) {
        const int e = (int)edge_base + i;
        const int s = src[e];
        const int d = dst[e];
        // subj gather: 4 halfs (8B) per lane
        const unsigned long long sv =
            *(const unsigned long long*)(subj + (size_t)s*HID + lane*4);
        *(unsigned long long*)&ein[i*EIN_STRIDE + lane*4] = sv;
        // rel = (loc[s]-loc[d]) @ trans_W + trans_b, 4 outputs per lane
        const float dx = loc[s*3+0] - loc[d*3+0];
        const float dy = loc[s*3+1] - loc[d*3+1];
        const float dz = loc[s*3+2] - loc[d*3+2];
        union { _Float16 h[4]; unsigned long long u; } p;
        #pragma unroll
        for (int j = 0; j < 4; ++j)
            p.h[j] = (_Float16)(tb[j] + dx*tw0[j] + dy*tw1[j] + dz*tw2[j]);
        *(unsigned long long*)&ein[i*EIN_STRIDE + HID + lane*4] = p.u;
        // obj gather
        const unsigned long long ov =
            *(const unsigned long long*)(obj + (size_t)d*HID + lane*4);
        *(unsigned long long*)&ein[i*EIN_STRIDE + 2*HID + lane*4] = ov;
    }

    const int m  = lane & 15;   // A row / C column
    const int kh = lane >> 4;   // K half-group selector

    // ---------------- layer 1: [EPB x 384] @ [384x256] + relu --------------
    for (int nt = 0; nt < 16; ++nt) {
        v8f acc[MTILES];
        #pragma unroll
        for (int mt = 0; mt < MTILES; ++mt) acc[mt] = (v8f){};
        for (int kt = 0; kt < 12; ++kt) {
            HalfVec B;
            const _Float16* bp = w1f + ((size_t)(kt*16 + nt)*512 + lane*16);
            B.q[0] = *(const f32x4*)(bp);
            B.q[1] = *(const f32x4*)(bp + 8);
            #pragma unroll
            for (int mt = 0; mt < MTILES; ++mt) {
                HalfVec A;
                const _Float16* a = &ein[(mt*16 + m)*EIN_STRIDE + kt*32 + kh*8];
                A.q[0] = *(const f32x4*)(a);
                A.q[1] = *(const f32x4*)(a + 16);
                acc[mt] = __builtin_amdgcn_wmma_f32_16x16x32_f16(
                    false, A.h, false, B.h, (short)0, acc[mt], false, false);
            }
        }
        const float bias = b1[nt*16 + m];
        #pragma unroll
        for (int mt = 0; mt < MTILES; ++mt) {
            #pragma unroll
            for (int r = 0; r < 8; ++r) {
                float x = acc[mt][r] + bias;
                x = x > 0.0f ? x : 0.0f;
                h1[(mt*16 + r + 8*kh)*H1_STRIDE + nt*16 + m] = (_Float16)x;
            }
        }
    }

    // ---------------- layer 2: [EPB x 256] @ [256x128] + relu --------------
    _Float16* h2 = ein;                      // edge_in no longer needed
    for (int nt = 0; nt < 8; ++nt) {
        v8f acc[MTILES];
        #pragma unroll
        for (int mt = 0; mt < MTILES; ++mt) acc[mt] = (v8f){};
        for (int kt = 0; kt < 8; ++kt) {
            HalfVec B;
            const _Float16* bp = w2f + ((size_t)(kt*8 + nt)*512 + lane*16);
            B.q[0] = *(const f32x4*)(bp);
            B.q[1] = *(const f32x4*)(bp + 8);
            #pragma unroll
            for (int mt = 0; mt < MTILES; ++mt) {
                HalfVec A;
                const _Float16* a = &h1[(mt*16 + m)*H1_STRIDE + kt*32 + kh*8];
                A.q[0] = *(const f32x4*)(a);
                A.q[1] = *(const f32x4*)(a + 16);
                acc[mt] = __builtin_amdgcn_wmma_f32_16x16x32_f16(
                    false, A.h, false, B.h, (short)0, acc[mt], false, false);
            }
        }
        const float bias = b2[nt*16 + m];
        #pragma unroll
        for (int mt = 0; mt < MTILES; ++mt) {
            #pragma unroll
            for (int r = 0; r < 8; ++r) {
                float x = acc[mt][r] + bias;
                x = x > 0.0f ? x : 0.0f;
                h2[(mt*16 + r + 8*kh)*H2_STRIDE + nt*16 + m] = (_Float16)x;
            }
        }
    }

    // ---------------- layer 3: [EPB x 128] @ [128x32(27)] -> scores --------
    for (int nt = 0; nt < 2; ++nt) {
        v8f acc[MTILES];
        #pragma unroll
        for (int mt = 0; mt < MTILES; ++mt) acc[mt] = (v8f){};
        #pragma unroll
        for (int kt = 0; kt < 4; ++kt) {
            HalfVec B;
            const _Float16* bp = w3f + ((size_t)(kt*2 + nt)*512 + lane*16);
            B.q[0] = *(const f32x4*)(bp);
            B.q[1] = *(const f32x4*)(bp + 8);
            #pragma unroll
            for (int mt = 0; mt < MTILES; ++mt) {
                HalfVec A;
                const _Float16* a = &h2[(mt*16 + m)*H2_STRIDE + kt*32 + kh*8];
                A.q[0] = *(const f32x4*)(a);
                A.q[1] = *(const f32x4*)(a + 16);
                acc[mt] = __builtin_amdgcn_wmma_f32_16x16x32_f16(
                    false, A.h, false, B.h, (short)0, acc[mt], false, false);
            }
        }
        const int n = nt*16 + m;
        if (n < NCLS) {
            const float bias = b3[n];
            #pragma unroll
            for (int mt = 0; mt < MTILES; ++mt) {
                #pragma unroll
                for (int r = 0; r < 8; ++r) {
                    const long row = edge_base + mt*16 + r + 8*kh;
                    scores[row*NCLS + n] = acc[mt][r] + bias;
                }
            }
        }
    }
}

// ---------------------------------------------------------------------------
extern "C" void kernel_launch(void* const* d_in, const int* in_sizes, int n_in,
                              void* d_out, int out_size, void* d_ws, size_t ws_size,
                              hipStream_t stream)
{
    (void)in_sizes; (void)n_in; (void)out_size; (void)ws_size;

    const int*   orientation  = (const int*)  d_in[0];
    const float* dimension    = (const float*)d_in[1];
    const float* location     = (const float*)d_in[2];
    const int*   src          = (const int*)  d_in[3];
    const int*   dst          = (const int*)  d_in[4];
    const int*   targets      = (const int*)  d_in[5];
    const float* orient_table = (const float*)d_in[6];
    const float* size_W       = (const float*)d_in[7];
    const float* size_b       = (const float*)d_in[8];
    const float* subj_W       = (const float*)d_in[9];
    const float* subj_b       = (const float*)d_in[10];
    const float* obj_W        = (const float*)d_in[11];
    const float* obj_b        = (const float*)d_in[12];
    const float* trans_W      = (const float*)d_in[13];
    const float* trans_b      = (const float*)d_in[14];
    const float* mlp_W1       = (const float*)d_in[15];
    const float* mlp_b1       = (const float*)d_in[16];
    const float* mlp_W2       = (const float*)d_in[17];
    const float* mlp_b2       = (const float*)d_in[18];
    const float* mlp_W3       = (const float*)d_in[19];
    const float* mlp_b3       = (const float*)d_in[20];

    // workspace carve-out (all 256B-aligned by construction)
    char* ws = (char*)d_ws;
    _Float16* subj_f16 = (_Float16*)ws;  ws += (size_t)NNODES*HID*sizeof(_Float16);
    _Float16* obj_f16  = (_Float16*)ws;  ws += (size_t)NNODES*HID*sizeof(_Float16);
    _Float16* w1f      = (_Float16*)ws;  ws += (size_t)192*512*sizeof(_Float16);
    _Float16* w2f      = (_Float16*)ws;  ws += (size_t)64*512*sizeof(_Float16);
    _Float16* w3f      = (_Float16*)ws;  ws += (size_t)8*512*sizeof(_Float16);

    node_embed_kernel<<<(NNODES + 127)/128, 128, 0, stream>>>(
        orientation, dimension, orient_table, size_W, size_b,
        subj_W, subj_b, obj_W, obj_b, subj_f16, obj_f16);

    repack_kernel<<<(192*512 + 255)/256, 256, 0, stream>>>(mlp_W1, w1f, 16, 2*HID, 192*512);
    repack_kernel<<<( 64*512 + 255)/256, 256, 0, stream>>>(mlp_W2, w2f,  8,   HID,  64*512);
    repack_kernel<<<(  8*512 + 255)/256, 256, 0, stream>>>(mlp_W3, w3f,  2,  NCLS,   8*512);

    const int nblocks = NEDGES / EPB;       // 15625 blocks, 64 edges each
    const size_t lds_bytes = (size_t)LDS_HALFS * sizeof(_Float16);   // 83968
    edge_mlp_kernel<<<nblocks, 32, lds_bytes, stream>>>(
        src, dst, location, trans_W, trans_b, subj_f16, obj_f16,
        w1f, mlp_b1, w2f, mlp_b2, w3f, mlp_b3, (float*)d_out);

    // tuple output: append targets (raw int32 bits) after the E*27 scores
    hipMemcpyAsync((char*)d_out + (size_t)NEDGES*NCLS*sizeof(float), targets,
                   (size_t)NEDGES*sizeof(int), hipMemcpyDeviceToDevice, stream);
}
